// HiLo_64544768524601
// MI455X (gfx1250) — matrix operand order repaired and checked by reading, not compiled, not run on hardware
//
#include <hip/hip_runtime.h>

// ---------------------------------------------------------------------------
// HiLo attention for MI455X (gfx1250): bf16 WMMA everywhere, flash attention.
// ---------------------------------------------------------------------------

typedef unsigned short bfraw;                                    // raw bf16
typedef __attribute__((ext_vector_type(16))) __bf16 bf16x16;     // WMMA A/B
typedef __attribute__((ext_vector_type(8)))  float  f32x8;       // WMMA C/D

union FragBF { uint4 q[2]; bf16x16 v; };

__device__ __forceinline__ bfraw f2bf(float f) {
  union { float f; unsigned u; } v; v.f = f;
  unsigned r = v.u + 0x7FFFu + ((v.u >> 16) & 1u);   // round-to-nearest-even
  return (bfraw)(r >> 16);
}
__device__ __forceinline__ float bf2f(bfraw h) {
  union { unsigned u; float f; } v; v.u = ((unsigned)h) << 16;
  return v.f;
}
__device__ __forceinline__ f32x8 wmma_bf16(bf16x16 a, bf16x16 b, f32x8 c) {
  // emits v_wmma_f32_16x16x32_bf16
  return __builtin_amdgcn_wmma_f32_16x16x32_bf16(false, a, false, b,
                                                 (short)0, c, false, false);
}

// ------------------------- elementwise prep kernels ------------------------

__global__ void pack_x_bf16(const float* __restrict__ x, bfraw* __restrict__ o,
                            int n) {
  int i = blockIdx.x * blockDim.x + threadIdx.x;
  if (i < n) o[i] = f2bf(x[i]);
}

// 2x2 avg pool of x [NT,512] (H=W=64) -> xp [NT/4,512] bf16
__global__ void pool_x_bf16(const float* __restrict__ x, bfraw* __restrict__ o,
                            int n /* = (NT/4)*512 */) {
  int i = blockIdx.x * blockDim.x + threadIdx.x;
  if (i >= n) return;
  int c  = i & 511;
  int bm = i >> 9;                 // pooled token
  int b  = bm >> 10;               // 1024 pooled tokens per image
  int m  = bm & 1023;
  int gy = m >> 5, gx = m & 31;
  size_t base = ((size_t)b * 4096 + (size_t)(2 * gy) * 64 + 2 * gx) * 512 + c;
  float s = x[base] + x[base + 512] + x[base + 64 * 512] + x[base + 65 * 512];
  o[i] = f2bf(s * 0.25f);
}

// weight [K,N] f32 -> dst [(colOff+n)][k] bf16 (column-major / transposed)
__global__ void pack_wT(const float* __restrict__ w, bfraw* __restrict__ dst,
                        int K, int N, int colOff) {
  int i = blockIdx.x * blockDim.x + threadIdx.x;
  if (i >= K * N) return;
  int n = i / K, k = i - n * K;
  dst[(size_t)(colOff + n) * K + k] = f2bf(w[(size_t)k * N + n]);
}

// hi1 = Y[:,0:256]; out = window-mean(2x2) - self, bf16 [NT,256]
__global__ void window_hi(const bfraw* __restrict__ Y, bfraw* __restrict__ o,
                          int n /* NT*256 */) {
  int i = blockIdx.x * blockDim.x + threadIdx.x;
  if (i >= n) return;
  int c = i & 255;
  int t = i >> 8;
  int r = t & 4095;
  int y = r >> 6, x = r & 63;
  int t00 = (t & ~4095) + ((y & ~1) << 6) + (x & ~1);
  size_t p = (size_t)t00 * 512 + c;
  float m = (bf2f(Y[p]) + bf2f(Y[p + 512]) + bf2f(Y[p + 64 * 512]) +
             bf2f(Y[p + 65 * 512])) * 0.25f;
  o[i] = f2bf(m - bf2f(Y[(size_t)t * 512 + c]));
}

// kv f32 [B*1024,512] -> Kbf [B*4][1024][64], Vt [B*4][64][1024] bf16
__global__ void pack_kv(const float* __restrict__ kv, bfraw* __restrict__ Kbf,
                        bfraw* __restrict__ Vt, int n /* rows*512 */) {
  int i = blockIdx.x * blockDim.x + threadIdx.x;
  if (i >= n) return;
  int c  = i & 511;
  int bm = i >> 9;
  int b  = bm >> 10, m = bm & 1023;
  float v = kv[i];
  if (c < 256) {
    int h = c >> 6, d = c & 63;
    Kbf[(((size_t)(b * 4 + h) * 1024) + m) * 64 + d] = f2bf(v);
  } else {
    int c2 = c - 256, h = c2 >> 6, d = c2 & 63;
    Vt[(((size_t)(b * 4 + h) * 64) + d) * 1024 + m] = f2bf(v);
  }
}

// ------------------------------ WMMA GEMM ---------------------------------
// C[M,N] = A[M,K] (bf16, row-major) x Wt[N,K] (bf16, pre-transposed weights)
// One 16x16 tile per wave; 4 waves/block tile N; K-loop step 32.
__global__ __launch_bounds__(128)
void gemm_bf16_wmma(const bfraw* __restrict__ A, int lda,
                    const bfraw* __restrict__ Wt, int K,
                    const float* __restrict__ bias,
                    float* __restrict__ outF, bfraw* __restrict__ outB,
                    int ldo, int colOff) {
  const int lane = threadIdx.x & 31;
  const int wave = threadIdx.x >> 5;
  const int half = lane >> 4;
  const int l16  = lane & 15;
  const int mbase = blockIdx.y * 16;
  const int nbase = (blockIdx.x * 4 + wave) * 16;

  f32x8 acc = {};
  const bfraw* arow = A  + (size_t)(mbase + l16) * lda;
  const bfraw* brow = Wt + (size_t)(nbase + l16) * K + 16 * half;

  for (int kb = 0; kb < K; kb += 32) {
    FragBF a, b;
    a.q[0] = *(const uint4*)(arow + kb + 16 * half);       // e0..7:  K=kb+16h+e
    a.q[1] = *(const uint4*)(arow + kb + 16 * half + 8);   // e8..15: K=kb+16h+e
    b.q[0] = *(const uint4*)(brow + kb);                   // e0..7:  K=kb+16h+e
    b.q[1] = *(const uint4*)(brow + kb + 8);
    if (kb + 32 < K) {
      __builtin_prefetch(arow + kb + 32, 0, 1);            // global_prefetch_b8
      __builtin_prefetch(brow + kb + 32, 0, 1);
    }
    acc = wmma_bf16(a.v, b.v, acc);
  }

  float bv = bias ? bias[nbase + l16] : 0.f;
  const int col = colOff + nbase + l16;
#pragma unroll
  for (int i = 0; i < 8; ++i) {
    int row = mbase + i + 8 * half;                        // C layout: M=i+8*half
    float v = acc[i] + bv;
    if (outF) outF[(size_t)row * ldo + col] = v;
    else      outB[(size_t)row * ldo + col] = f2bf(v);
  }
}

// --------------------------- flash attention ------------------------------
// Per wave: 16 queries, online softmax over 1024 keys in 32-key chunks.
// scores/probs never touch global memory (avoids ~1 GB of traffic).
__global__ __launch_bounds__(128)
void hilo_attn(const bfraw* __restrict__ Y,    // [NT,512], q at cols 256..511
               const bfraw* __restrict__ Kbf,  // [B*4][1024][64]
               const bfraw* __restrict__ Vt,   // [B*4][64][1024]
               bfraw* __restrict__ att) {      // [NT,256]
  __shared__ __align__(16) bfraw Plds[4][16][32];
  const float SCALE = 0.125f;                  // 64^-0.5
  const int lane = threadIdx.x & 31;
  const int wave = threadIdx.x >> 5;
  const int half = lane >> 4;
  const int l16  = lane & 15;
  const int gw = blockIdx.x * 4 + wave;
  const int qt = gw & 255;                     // 256 q-tiles per (b,h)
  const int bh = gw >> 8;
  const int b  = bh >> 2, h = bh & 3;

  // Q fragments (16 rows x 64 head-dims -> two K=32 A-frags), loaded once
  const bfraw* qrow = Y + (size_t)(b * 4096 + qt * 16 + l16) * 512 + 256 + h * 64;
  FragBF qf0, qf1;
  qf0.q[0] = *(const uint4*)(qrow + 16 * half);
  qf0.q[1] = *(const uint4*)(qrow + 8 + 16 * half);
  qf1.q[0] = *(const uint4*)(qrow + 32 + 16 * half);
  qf1.q[1] = *(const uint4*)(qrow + 40 + 16 * half);

  const bfraw* kbase = Kbf + (size_t)bh * 1024 * 64;
  const bfraw* vbase = Vt  + (size_t)bh * 64 * 1024;

  f32x8 o0 = {}, o1 = {}, o2 = {}, o3 = {};
  float mrun[8], lrun[8];
#pragma unroll
  for (int i = 0; i < 8; ++i) { mrun[i] = -3.0e38f; lrun[i] = 0.f; }

  for (int kc = 0; kc < 1024; kc += 32) {
    // S = Q @ K^T for 32 keys (two 16x16 tiles), contraction over d=64
    f32x8 s[2];
#pragma unroll
    for (int kh = 0; kh < 2; ++kh) {
      const bfraw* krow = kbase + (size_t)(kc + 16 * kh + l16) * 64 + 16 * half;
      FragBF bk0, bk1;
      bk0.q[0] = *(const uint4*)(krow);        // d = 16*half + e
      bk0.q[1] = *(const uint4*)(krow + 8);
      bk1.q[0] = *(const uint4*)(krow + 32);   // d = 32 + 16*half + e
      bk1.q[1] = *(const uint4*)(krow + 40);
      f32x8 c = {};
      c = wmma_bf16(qf0.v, bk0.v, c);
      c = wmma_bf16(qf1.v, bk1.v, c);
      s[kh] = c;
    }

    // Online softmax. C layout: lane = key column, VGPR i = row (i + 8*half),
    // so row reductions are cross-lane within each 16-lane group.
#pragma unroll
    for (int i = 0; i < 8; ++i) {
      float sa = s[0][i] * SCALE, sb = s[1][i] * SCALE;
      float t = fmaxf(sa, sb);
      t = fmaxf(t, __shfl_xor(t, 1, 16));
      t = fmaxf(t, __shfl_xor(t, 2, 16));
      t = fmaxf(t, __shfl_xor(t, 4, 16));
      t = fmaxf(t, __shfl_xor(t, 8, 16));
      float mn = fmaxf(mrun[i], t);
      float pa = __expf(sa - mn), pb = __expf(sb - mn);
      float rs = pa + pb;
      rs += __shfl_xor(rs, 1, 16);
      rs += __shfl_xor(rs, 2, 16);
      rs += __shfl_xor(rs, 4, 16);
      rs += __shfl_xor(rs, 8, 16);
      float corr = __expf(mrun[i] - mn);
      lrun[i] = lrun[i] * corr + rs;
      mrun[i] = mn;
      o0[i] *= corr; o1[i] *= corr; o2[i] *= corr; o3[i] *= corr;
      Plds[wave][i + 8 * half][l16]      = f2bf(pa);   // keys kc+0..15
      Plds[wave][i + 8 * half][16 + l16] = f2bf(pb);   // keys kc+16..31
    }
    // Wave-private LDS region: only need this wave's DS ops complete before
    // the cross-lane re-read (no block barrier required on CDNA5).
    asm volatile("s_wait_dscnt 0" ::: "memory");

    // P (16x32) as A-frag, transposed through LDS
    FragBF pf;
    pf.q[0] = *(const uint4*)(&Plds[wave][l16][16 * half]);
    pf.q[1] = *(const uint4*)(&Plds[wave][l16][16 * half + 8]);

    // O += P @ V : four 16-dim output tiles, contraction over 32 keys
    FragBF vf;
    const bfraw* vrow = vbase + (size_t)l16 * 1024 + kc + 16 * half;
    vf.q[0] = *(const uint4*)(vrow);
    vf.q[1] = *(const uint4*)(vrow + 8);
    o0 = wmma_bf16(pf.v, vf.v, o0);
    vrow += 16 * 1024;
    vf.q[0] = *(const uint4*)(vrow); vf.q[1] = *(const uint4*)(vrow + 8);
    o1 = wmma_bf16(pf.v, vf.v, o1);
    vrow += 16 * 1024;
    vf.q[0] = *(const uint4*)(vrow); vf.q[1] = *(const uint4*)(vrow + 8);
    o2 = wmma_bf16(pf.v, vf.v, o2);
    vrow += 16 * 1024;
    vf.q[0] = *(const uint4*)(vrow); vf.q[1] = *(const uint4*)(vrow + 8);
    o3 = wmma_bf16(pf.v, vf.v, o3);
  }

#pragma unroll
  for (int i = 0; i < 8; ++i) {
    float inv = 1.f / lrun[i];
    size_t row = (size_t)(b * 4096 + qt * 16 + i + 8 * half) * 256 + h * 64 + l16;
    att[row]      = f2bf(o0[i] * inv);
    att[row + 16] = f2bf(o1[i] * inv);
    att[row + 32] = f2bf(o2[i] * inv);
    att[row + 48] = f2bf(o3[i] * inv);
  }
}

// ------------------------------- host side --------------------------------

extern "C" void kernel_launch(void* const* d_in, const int* in_sizes, int n_in,
                              void* d_out, int out_size, void* d_ws, size_t ws_size,
                              hipStream_t stream) {
  const float* x        = (const float*)d_in[0];
  const float* l_q_w    = (const float*)d_in[2];
  const float* l_kv_w   = (const float*)d_in[3];
  const float* l_proj_w = (const float*)d_in[4];
  const float* l_proj_b = (const float*)d_in[5];
  const float* h_qkv_w  = (const float*)d_in[6];
  const float* h_proj_w = (const float*)d_in[7];
  const float* h_proj_b = (const float*)d_in[8];

  const int NT = in_sizes[0] / 512;   // total tokens (B*4096), 32768 for B=8
  const int B  = NT / 4096;
  const int NP = NT / 4;              // pooled tokens

  char* p = (char*)d_ws;
  size_t off = 0;
  auto alloc = [&](size_t bytes) {
    char* r = p + off;
    off += (bytes + 255) & ~(size_t)255;
    return r;
  };
  bfraw* xbf  = (bfraw*)alloc((size_t)NT * 512 * 2);
  bfraw* xpbf = (bfraw*)alloc((size_t)NP * 512 * 2);
  bfraw* Wt1  = (bfraw*)alloc(512 * 512 * 2);   // [hi_qkv | l_q] transposed
  bfraw* Wt2  = (bfraw*)alloc(512 * 512 * 2);   // l_kv transposed
  bfraw* Wt3  = (bfraw*)alloc(256 * 256 * 2);   // h_proj transposed
  bfraw* Wt4  = (bfraw*)alloc(256 * 256 * 2);   // l_proj transposed
  bfraw* Ybf  = (bfraw*)alloc((size_t)NT * 512 * 2);   // [hi1 | q]
  bfraw* hibf = (bfraw*)alloc((size_t)NT * 256 * 2);
  float* kvf  = (float*)alloc((size_t)NP * 512 * 4);
  bfraw* Kbf  = (bfraw*)alloc((size_t)B * 4 * 1024 * 64 * 2);
  bfraw* Vt   = (bfraw*)alloc((size_t)B * 4 * 64 * 1024 * 2);
  bfraw* attb = (bfraw*)alloc((size_t)NT * 256 * 2);
  float* out  = (float*)d_out;

  const int TB = 256;
  // 1) bf16 pack + pool + weight transposes
  pack_x_bf16<<<(NT * 512 + TB - 1) / TB, TB, 0, stream>>>(x, xbf, NT * 512);
  pool_x_bf16<<<(NP * 512 + TB - 1) / TB, TB, 0, stream>>>(x, xpbf, NP * 512);
  pack_wT<<<(512 * 256 + TB - 1) / TB, TB, 0, stream>>>(h_qkv_w,  Wt1, 512, 256, 0);
  pack_wT<<<(512 * 256 + TB - 1) / TB, TB, 0, stream>>>(l_q_w,    Wt1, 512, 256, 256);
  pack_wT<<<(512 * 512 + TB - 1) / TB, TB, 0, stream>>>(l_kv_w,   Wt2, 512, 512, 0);
  pack_wT<<<(256 * 256 + TB - 1) / TB, TB, 0, stream>>>(h_proj_w, Wt3, 256, 256, 0);
  pack_wT<<<(256 * 256 + TB - 1) / TB, TB, 0, stream>>>(l_proj_w, Wt4, 256, 256, 0);

  // 2) fused input GEMM: Y = x @ [h_qkv_w | l_q_w]   (M=NT, K=512, N=512)
  gemm_bf16_wmma<<<dim3(512 / 64, NT / 16), 128, 0, stream>>>(
      xbf, 512, Wt1, 512, nullptr, nullptr, Ybf, 512, 0);

  // 3) kv GEMM: kv = xp @ l_kv_w   (M=NP, K=512, N=512) -> f32
  gemm_bf16_wmma<<<dim3(512 / 64, NP / 16), 128, 0, stream>>>(
      xpbf, 512, Wt2, 512, nullptr, kvf, nullptr, 512, 0);

  // 4) hi window op (mean - self) on Y[:,0:256]
  window_hi<<<(NT * 256 + TB - 1) / TB, TB, 0, stream>>>(Ybf, hibf, NT * 256);

  // 5) split/pack K,V (V transposed for contiguous-key B-frags)
  pack_kv<<<(NP * 512 + TB - 1) / TB, TB, 0, stream>>>(kvf, Kbf, Vt, NP * 512);

  // 6) flash attention: (B*4) heads x 256 q-tiles, 4 waves/block
  hilo_attn<<<B * 256, 128, 0, stream>>>(Ybf, Kbf, Vt, attb);

  // 7) projections straight into concatenated f32 output
  gemm_bf16_wmma<<<dim3(256 / 64, NT / 16), 128, 0, stream>>>(
      hibf, 256, Wt3, 256, h_proj_b, out, nullptr, 512, 0);
  gemm_bf16_wmma<<<dim3(256 / 64, NT / 16), 128, 0, stream>>>(
      attb, 256, Wt4, 256, l_proj_b, out, nullptr, 512, 256);
}
